// Ftree2fpGAT_3788161155358
// MI455X (gfx1250) — matrix operand order repaired and checked by reading, not compiled, not run on hardware
//
#include <hip/hip_runtime.h>
#include <hip/hip_bf16.h>
#include <math.h>

// ---------------- problem constants (from reference) ----------------
#define NN   50000      // nodes
#define NE   800000     // edges
#define NEF  850000     // edges + self loops
#define FD   128        // feat = hidden
#define NH   4          // heads
#define HC   32         // channels/head
#define NG   1000       // graphs
#define OUTD 2048
#define MP   1008       // pooled rows padded to multiple of 16 (63 tiles)
#define LEAKY 0.2f

typedef __attribute__((ext_vector_type(16))) __bf16 v16bf;
typedef __attribute__((ext_vector_type(8)))  __bf16 v8bf;
typedef __attribute__((ext_vector_type(8)))  float  v8f;

// ---------------- helpers ----------------
__device__ __forceinline__ __hip_bfloat16 f2bf(float f) { return __float2bfloat16(f); }

// IEEE float atomic max via int/uint ordering trick (works for mixed signs, -inf init)
__device__ __forceinline__ void atomicMaxF(float* addr, float val) {
    if (val >= 0.0f) atomicMax((int*)addr, __float_as_int(val));
    else             atomicMin((unsigned int*)addr, __float_as_uint(val));
}

// ---------------- generic fill ----------------
__global__ void fill_f(float* __restrict__ p, int n, float v) {
    int i = blockIdx.x * blockDim.x + threadIdx.x;
    if (i < n) p[i] = v;
}

__global__ void conv_bf16(const float* __restrict__ x, __hip_bfloat16* __restrict__ xb, int n) {
    int i = blockIdx.x * blockDim.x + threadIdx.x;
    if (i < n) xb[i] = f2bf(x[i]);
}

// ---------------- self-loop edge_attr (fill_value='mean') ----------------
__global__ void edge_count_sum(const int* __restrict__ dst, const float* __restrict__ ea,
                               float* __restrict__ cnt, float* __restrict__ easum) {
    int e = blockIdx.x * blockDim.x + threadIdx.x;
    if (e < NE) {
        int d = dst[e];
        atomicAdd(&cnt[d], 1.0f);
        atomicAdd(&easum[d], ea[e]);
    }
}
__global__ void ea_self_div(float* __restrict__ easum, const float* __restrict__ cnt) {
    int i = blockIdx.x * blockDim.x + threadIdx.x;
    if (i < NN) easum[i] = (cnt[i] > 0.0f) ? (easum[i] / cnt[i]) : 0.0f;
}

// ---------------- pack W[K][ldn] -> WMMA bf16 B-fragment order ----------------
// Fragment (ntile, s=K/32 step, lane): element i holds K = s*32 + (lane>=16?16:0) + i,
// column n = ntile*16 + (lane&15).  Stored contiguously: 16 bf16 (32B) per lane.
__global__ void pack_B(const float* __restrict__ W, __hip_bfloat16* __restrict__ Bp,
                       int ldn, int nTiles) {
    int t = blockIdx.x * blockDim.x + threadIdx.x;
    if (t >= nTiles * 4 * 32) return;
    int lane  = t & 31;
    int s     = (t >> 5) & 3;
    int ntile = t >> 7;
    int col   = ntile * 16 + (lane & 15);
    int kb    = s * 32 + ((lane & 16) ? 16 : 0);
    __hip_bfloat16* o = Bp + (size_t)t * 16;
#pragma unroll
    for (int i = 0; i < 16; ++i) o[i] = f2bf(W[(size_t)(kb + i) * ldn + col]);
}

// ---------------- A fragment load (16-bit A 16x32, contiguous k order) ----------------
// lane<16: rows M=lane, K = 0..15; lane>=16: K = 16..31 (contiguous, matches pack_B).
__device__ __forceinline__ v16bf load_a_frag(const __hip_bfloat16* __restrict__ arow, int lane) {
    int base = (lane & 16) ? 16 : 0;
    v16bf a = *(const v16bf*)(const void*)(arow + base);
    return a;
}

// ---------------- WMMA GEMM: C[M][Ncols] = A[M][128] * B  (bf16 in, f32 acc) ----
// one wave per 16x16 tile; block = 8 waves covering 8 consecutive n-tiles.
__global__ void gemm_bf16_wmma(const __hip_bfloat16* __restrict__ A,
                               const __hip_bfloat16* __restrict__ Bp,
                               float* __restrict__ C, int Ncols) {
    int lane  = threadIdx.x & 31;
    int wave  = threadIdx.x >> 5;
    int mtile = blockIdx.x;
    int ntile = blockIdx.y * 8 + wave;
    const __hip_bfloat16* arow = A + (size_t)(mtile * 16 + (lane & 15)) * 128;
    v8f acc = {};
#pragma unroll
    for (int s = 0; s < 4; ++s) {
        v16bf a = load_a_frag(arow + s * 32, lane);
        v16bf b = *(const v16bf*)(const void*)(Bp + ((size_t)(ntile * 4 + s) * 32 + lane) * 16);
        acc = __builtin_amdgcn_wmma_f32_16x16x32_bf16(false, a, false, b, (short)0, acc, false, false);
    }
    int rbase = mtile * 16 + ((lane & 16) ? 8 : 0);
    int col   = ntile * 16 + (lane & 15);
#pragma unroll
    for (int v = 0; v < 8; ++v) C[(size_t)(rbase + v) * Ncols + col] = acc[v];
}

// WMMA GEMM with fused bias + sigmoid epilogue (classifier head)
__global__ void gemm_bf16_wmma_sig(const __hip_bfloat16* __restrict__ A,
                                   const __hip_bfloat16* __restrict__ Bp,
                                   const float* __restrict__ bias,
                                   float* __restrict__ C, int Ncols, int Mreal) {
    int lane  = threadIdx.x & 31;
    int wave  = threadIdx.x >> 5;
    int mtile = blockIdx.x;
    int ntile = blockIdx.y * 8 + wave;
    const __hip_bfloat16* arow = A + (size_t)(mtile * 16 + (lane & 15)) * 128;
    v8f acc = {};
#pragma unroll
    for (int s = 0; s < 4; ++s) {
        v16bf a = load_a_frag(arow + s * 32, lane);
        v16bf b = *(const v16bf*)(const void*)(Bp + ((size_t)(ntile * 4 + s) * 32 + lane) * 16);
        acc = __builtin_amdgcn_wmma_f32_16x16x32_bf16(false, a, false, b, (short)0, acc, false, false);
    }
    int rbase = mtile * 16 + ((lane & 16) ? 8 : 0);
    int col   = ntile * 16 + (lane & 15);
#pragma unroll
    for (int v = 0; v < 8; ++v) {
        int r = rbase + v;
        if (r < Mreal) {
            float z = acc[v] + bias[col];
            C[(size_t)r * Ncols + col] = 1.0f / (1.0f + __expf(-z));
        }
    }
}

// ---------------- attention scalars ----------------
// ce[h] = sum_c We[h*32+c] * att_e[h*32+c]   (al_e[edge,h] = ea * ce[h])
__global__ void ce_kernel(const float* __restrict__ We_l, const float* __restrict__ ae_l,
                          float* __restrict__ ce) {
    int h = threadIdx.x;
    if (h < NH) {
        float s = 0.0f;
#pragma unroll
        for (int c = 0; c < HC; ++c) s += We_l[h * HC + c] * ae_l[h * HC + c];
        ce[h] = s;
    }
}

// als[n,h] = h[n,h,:].a_src[h,:], ald likewise
__global__ void als_ald(const float* __restrict__ hx, const float* __restrict__ a_s,
                        const float* __restrict__ a_d, float* __restrict__ als,
                        float* __restrict__ ald) {
    int t = blockIdx.x * blockDim.x + threadIdx.x;
    if (t >= NN * NH) return;
    int n = t >> 2, hh = t & 3;
    const float* hp = hx + (size_t)n * FD + hh * HC;
    const float* as = a_s + hh * HC;
    const float* ad = a_d + hh * HC;
    float s1 = 0.0f, s2 = 0.0f;
#pragma unroll
    for (int c = 0; c < HC; ++c) { s1 += hp[c] * as[c]; s2 += hp[c] * ad[c]; }
    als[t] = s1; ald[t] = s2;
}

// ---------------- edge passes ----------------
__global__ void edge_logits(const int* __restrict__ src, const int* __restrict__ dst,
                            const float* __restrict__ ea, const float* __restrict__ eaself,
                            const float* __restrict__ als, const float* __restrict__ ald,
                            const float* __restrict__ ce, float* __restrict__ m,
                            float* __restrict__ elog) {
    int e = blockIdx.x * blockDim.x + threadIdx.x;
    if (e >= NEF) return;
    int s, d; float eav;
    if (e < NE) { s = src[e]; d = dst[e]; eav = ea[e]; }
    else        { s = e - NE; d = s;      eav = eaself[s]; }
#pragma unroll
    for (int h = 0; h < NH; ++h) {
        float l = als[s * NH + h] + ald[d * NH + h] + eav * ce[h];
        l = (l > 0.0f) ? l : (LEAKY * l);            // leaky_relu(0.2)
        elog[(size_t)e * NH + h] = l;
        atomicMaxF(&m[d * NH + h], l);
    }
}

__global__ void edge_exp(const int* __restrict__ dst, float* __restrict__ elog,
                         const float* __restrict__ m, float* __restrict__ den) {
    int e = blockIdx.x * blockDim.x + threadIdx.x;
    if (e >= NEF) return;
    int d = (e < NE) ? dst[e] : (e - NE);
#pragma unroll
    for (int h = 0; h < NH; ++h) {
        float ex = __expf(elog[(size_t)e * NH + h] - m[d * NH + h]);
        elog[(size_t)e * NH + h] = ex;               // store exp in place
        atomicAdd(&den[d * NH + h], ex);
    }
}

// xnext[dst,c] += (ex / (den[dst,h]+1e-16)) * h[src,c]    (thread per edge-channel)
__global__ void edge_msg(const int* __restrict__ src, const int* __restrict__ dst,
                         const float* __restrict__ elog, const float* __restrict__ den,
                         const float* __restrict__ hx, float* __restrict__ xnext) {
    long long t = (long long)blockIdx.x * blockDim.x + threadIdx.x;
    if (t >= (long long)NEF * FD) return;
    int c = (int)(t & (FD - 1));
    int e = (int)(t >> 7);
    int s, d;
    if (e < NE) { s = src[e]; d = dst[e]; } else { s = e - NE; d = s; }
    int hh = c >> 5;
    float alpha = elog[(size_t)e * NH + hh] / (den[d * NH + hh] + 1e-16f);
    atomicAdd(&xnext[(size_t)d * FD + c], alpha * hx[(size_t)s * FD + c]);
}

__global__ void relu_bias(float* __restrict__ x, const float* __restrict__ b, int n) {
    int i = blockIdx.x * blockDim.x + threadIdx.x;
    if (i < n) { float v = x[i] + b[i & (FD - 1)]; x[i] = v > 0.0f ? v : 0.0f; }
}

// ---------------- pooling ----------------
__global__ void pool_init(float* __restrict__ pooled) {
    int i = blockIdx.x * blockDim.x + threadIdx.x;
    if (i < MP * FD) pooled[i] = (i < NG * FD) ? -INFINITY : 0.0f;  // pad rows -> 0
}
__global__ void pool_max(const float* __restrict__ x, const int* __restrict__ batch,
                         float* __restrict__ pooled) {
    int t = blockIdx.x * blockDim.x + threadIdx.x;
    if (t >= NN * FD) return;
    int n = t >> 7, c = t & (FD - 1);
    atomicMaxF(&pooled[(size_t)batch[n] * FD + c], x[t]);
}

// ---------------- launcher ----------------
extern "C" void kernel_launch(void* const* d_in, const int* in_sizes, int n_in,
                              void* d_out, int out_size, void* d_ws, size_t ws_size,
                              hipStream_t stream) {
    const float* x_in   = (const float*)d_in[0];
    const int*   eidx   = (const int*)d_in[1];
    const float* eattr  = (const float*)d_in[2];
    const int*   batch  = (const int*)d_in[3];
    const float* Ws     = (const float*)d_in[4];
    const float* att_s  = (const float*)d_in[5];
    const float* att_d  = (const float*)d_in[6];
    const float* We     = (const float*)d_in[7];
    const float* att_e  = (const float*)d_in[8];
    const float* biases = (const float*)d_in[9];
    const float* lin1_w = (const float*)d_in[10];
    const float* lin1_b = (const float*)d_in[11];
    float* out = (float*)d_out;
    const int* src = eidx;
    const int* dst = eidx + NE;

    // workspace carve-out (256B aligned)
    char* ws = (char*)d_ws;
    size_t off = 0;
    auto take = [&](size_t bytes) -> void* {
        void* p = ws + off; off += (bytes + 255) & ~(size_t)255; return p;
    };
    float* cnt     = (float*)take((size_t)NN * 4);
    float* eaself  = (float*)take((size_t)NN * 4);              // sum -> mean in place
    float* ce      = (float*)take(256);
    __hip_bfloat16* Wb   = (__hip_bfloat16*)take((size_t)8 * 4 * 32 * 16 * 2);
    __hip_bfloat16* xb   = (__hip_bfloat16*)take((size_t)NN * FD * 2);
    float* hbuf    = (float*)take((size_t)NN * FD * 4);
    float* xnext   = (float*)take((size_t)NN * FD * 4);
    float* als     = (float*)take((size_t)NN * NH * 4);
    float* ald     = (float*)take((size_t)NN * NH * 4);
    float* mbuf    = (float*)take((size_t)NN * NH * 4);
    float* den     = (float*)take((size_t)NN * NH * 4);
    float* elog    = (float*)take((size_t)NEF * NH * 4);
    float* pooled  = (float*)take((size_t)MP * FD * 4);
    __hip_bfloat16* pooledb = (__hip_bfloat16*)take((size_t)MP * FD * 2);
    __hip_bfloat16* l1b     = (__hip_bfloat16*)take((size_t)128 * 4 * 32 * 16 * 2);

    const int B = 256;
    auto nb = [](long long n, int b) { return (unsigned)((n + b - 1) / b); };

    // self-loop edge_attr = mean over incoming edges
    fill_f<<<nb(NN, B), B, 0, stream>>>(cnt, NN, 0.0f);
    fill_f<<<nb(NN, B), B, 0, stream>>>(eaself, NN, 0.0f);
    edge_count_sum<<<nb(NE, B), B, 0, stream>>>(dst, eattr, cnt, eaself);
    ea_self_div<<<nb(NN, B), B, 0, stream>>>(eaself, cnt);

    for (int l = 0; l < 3; ++l) {
        const float* xcur = (l == 0) ? x_in : xnext;
        pack_B<<<nb(8 * 4 * 32, B), B, 0, stream>>>(Ws + (size_t)l * FD * FD, Wb, FD, 8);
        ce_kernel<<<1, 32, 0, stream>>>(We + (size_t)l * FD, att_e + (size_t)l * FD, ce);
        conv_bf16<<<nb((long long)NN * FD, B), B, 0, stream>>>(xcur, xb, NN * FD);

        dim3 g1(NN / 16, 1);                       // 3125 m-tiles x 8 n-tiles (8 waves)
        gemm_bf16_wmma<<<g1, 256, 0, stream>>>(xb, Wb, hbuf, FD);

        als_ald<<<nb((long long)NN * NH, B), B, 0, stream>>>(
            hbuf, att_s + (size_t)l * FD, att_d + (size_t)l * FD, als, ald);

        fill_f<<<nb(NN * NH, B), B, 0, stream>>>(mbuf, NN * NH, -INFINITY);
        fill_f<<<nb(NN * NH, B), B, 0, stream>>>(den, NN * NH, 0.0f);
        fill_f<<<nb((long long)NN * FD, B), B, 0, stream>>>(xnext, NN * FD, 0.0f);

        edge_logits<<<nb(NEF, B), B, 0, stream>>>(src, dst, eattr, eaself, als, ald, ce, mbuf, elog);
        edge_exp<<<nb(NEF, B), B, 0, stream>>>(dst, elog, mbuf, den);
        edge_msg<<<nb((long long)NEF * FD, B), B, 0, stream>>>(src, dst, elog, den, hbuf, xnext);
        relu_bias<<<nb((long long)NN * FD, B), B, 0, stream>>>(xnext, biases + (size_t)l * FD, NN * FD);
    }

    // graph max-pool (padded to 1008 rows) -> bf16 -> WMMA head with bias+sigmoid
    pool_init<<<nb(MP * FD, B), B, 0, stream>>>(pooled);
    pool_max<<<nb((long long)NN * FD, B), B, 0, stream>>>(xnext, batch, pooled);
    conv_bf16<<<nb(MP * FD, B), B, 0, stream>>>(pooled, pooledb, MP * FD);
    pack_B<<<nb(128 * 4 * 32, B), B, 0, stream>>>(lin1_w, l1b, OUTD, 128);

    dim3 g2(MP / 16, OUTD / 128);                  // 63 x 16 blocks, 8 waves each
    gemm_bf16_wmma_sig<<<g2, 256, 0, stream>>>(pooledb, l1b, lin1_b, out, OUTD, NG);
}
